// DilatedAttention_28140625723862
// MI455X (gfx1250) — compile-verified, matrix operand order checked by reading, not run on hardware
//
#include <hip/hip_runtime.h>

typedef _Float16 h16;
typedef __attribute__((ext_vector_type(16))) _Float16 v16h;
typedef __attribute__((ext_vector_type(8)))  _Float16 h16x8;
typedef __attribute__((ext_vector_type(8)))  float    v8f;

#define D_MODEL   1024
#define NUM_HEADS 16
#define HEAD_DIM  64
#define DIL       4
#define SEG       512
#define BATCH     4
#define SEQ       8192
#define SD        (SEQ / DIL)   /* 2048 dilated tokens per (branch,batch) */

// ---------------------------------------------------------------------------
// WMMA helpers
// ---------------------------------------------------------------------------
__device__ __forceinline__ v8f wmma_f16(v16h a, v16h b, v8f c) {
  // D = A(16x32 f16) * B(32x16 f16) + C(16x16 f32)
  return __builtin_amdgcn_wmma_f32_16x16x32_f16(false, a, false, b, (short)0, c,
                                                false, false);
}

// Load a 16-wide f16 fragment for one lane. rowptr already includes the
// per-lane row (row = lane%16). ISA layout for 16-bit A/B frags:
//   vgprs 0..3 : K = 8*(lane/16) + 0..7      (contiguous)
//   vgprs 4..7 : K = 16 + 8*(lane/16) + 0..7 (contiguous)
__device__ __forceinline__ v16h ldfrag(const h16* rowptr, int k0, int half) {
  union { v16h v; h16x8 h[2]; } u;
  const h16* p = rowptr + k0 + half * 8;
  u.h[0] = *(const h16x8*)(p);
  u.h[1] = *(const h16x8*)(p + 16);
  return u.v;
}

// ---------------------------------------------------------------------------
// Kernel 1: fp32 -> f16 convert (grid-stride)
// ---------------------------------------------------------------------------
__global__ void k_cvt(const float* __restrict__ src, h16* __restrict__ dst,
                      long n) {
  long i = (long)blockIdx.x * blockDim.x + threadIdx.x;
  long stride = (long)gridDim.x * blockDim.x;
  for (; i < n; i += stride) dst[i] = (h16)src[i];
}

// ---------------------------------------------------------------------------
// Kernel 2: QKV projection. One wave = 32x64 output tile; the 8 waves of a
// block share the same 64 N-columns (identical B-fragment addresses -> WGP$
// hits) and cover 8 consecutive 32-row M-tiles.
//   M = 2048 tokens (per branch,batch), N = 3072 features, K = 1024.
// Q is pre-scaled by 1/sqrt(64); V is stored transposed [head][d][token].
// ---------------------------------------------------------------------------
__global__ __launch_bounds__(256) void k_qkv(const h16* __restrict__ xbf,
                                             const h16* __restrict__ wq,
                                             const float* __restrict__ bqkv,
                                             h16* __restrict__ Q,
                                             h16* __restrict__ K,
                                             h16* __restrict__ Vt) {
  const int lane = threadIdx.x & 31;
  const int l15  = lane & 15;
  const int half = lane >> 4;
  const int wib  = threadIdx.x >> 5;
  int gb = blockIdx.x;                      // 48 * 8 * 4 * 4 = 6144 blocks
  const int nt   = gb % 48; gb /= 48;       // 3072/64 column strips
  const int mblk = gb % 8;  gb /= 8;        // 8 m-blocks of 8x32 rows
  const int b    = gb % 4;
  const int br   = gb / 4;
  const int m0 = (mblk * 8 + wib) * 32;
  const int n0 = nt * 64;

  const h16* Arow[2];
  Arow[0] = xbf + ((size_t)b * SEQ + br + 4 * (m0 + l15)) * D_MODEL;
  Arow[1] = xbf + ((size_t)b * SEQ + br + 4 * (m0 + 16 + l15)) * D_MODEL;
  const h16* W = wq + (size_t)br * 3 * D_MODEL * D_MODEL;
  const h16* Brow[4];
#pragma unroll
  for (int t = 0; t < 4; ++t)
    Brow[t] = W + (size_t)(n0 + t * 16 + l15) * D_MODEL;

  v8f acc[2][4] = {};
  for (int k = 0; k < D_MODEL; k += 32) {
    const v16h a0 = ldfrag(Arow[0], k, half);
    const v16h a1 = ldfrag(Arow[1], k, half);
#pragma unroll
    for (int t = 0; t < 4; ++t) {
      const v16h bt = ldfrag(Brow[t], k, half);
      acc[0][t] = wmma_f16(a0, bt, acc[0][t]);
      acc[1][t] = wmma_f16(a1, bt, acc[1][t]);
    }
  }

  const int bb = br * 4 + b;
#pragma unroll
  for (int q = 0; q < 2; ++q)
#pragma unroll
    for (int t = 0; t < 4; ++t) {
      const int e = n0 + t * 16 + l15;
      const float bias = bqkv[br * 3 * D_MODEL + e];
#pragma unroll
      for (int r = 0; r < 8; ++r) {
        const int j = m0 + q * 16 + r + 8 * half;
        const float v = acc[q][t][r] + bias;
        if (e < D_MODEL) {                     // Q (pre-scaled by 1/sqrt(hd))
          const int h = e >> 6, d = e & 63;
          Q[(((size_t)bb * NUM_HEADS + h) * SD + j) * HEAD_DIM + d] =
              (h16)(v * 0.125f);
        } else if (e < 2 * D_MODEL) {          // K
          const int e2 = e - D_MODEL, h = e2 >> 6, d = e2 & 63;
          K[(((size_t)bb * NUM_HEADS + h) * SD + j) * HEAD_DIM + d] = (h16)v;
        } else {                               // V, transposed
          const int e3 = e - 2 * D_MODEL, h = e3 >> 6, d = e3 & 63;
          Vt[(((size_t)bb * NUM_HEADS + h) * HEAD_DIM + d) * SD + j] = (h16)v;
        }
      }
    }
}

// ---------------------------------------------------------------------------
// Kernel 3: segmented flash attention. One wave = 32 query rows; the 8 waves
// of a block share one (bb,head,segment) so K/V fragment loads are identical
// across waves (WGP$ reuse). Online softmax over 512 keys in 32-key chunks.
// ---------------------------------------------------------------------------
__global__ __launch_bounds__(256) void k_attn(const h16* __restrict__ Q,
                                              const h16* __restrict__ K,
                                              const h16* __restrict__ Vt,
                                              h16* __restrict__ O) {
  __shared__ __align__(16) h16 lds[8][32 * 32];   // per-wave P staging (2 KB)
  const int lane = threadIdx.x & 31;
  const int l15  = lane & 15;
  const int half = lane >> 4;
  const int wib  = threadIdx.x >> 5;
  int gb = blockIdx.x;                  // 2 * 4 * 16 * 16 = 2048 blocks
  const int qb  = gb % 2;  gb /= 2;     // 16 q-tiles of 32 rows -> 2 blocks
  const int seg = gb % 4;  gb /= 4;
  const int h   = gb % 16;
  const int bb  = gb / 16;
  const int j0  = seg * SEG + (qb * 8 + wib) * 32;

  const h16* Qh = Q  + ((size_t)bb * NUM_HEADS + h) * SD * HEAD_DIM;
  const h16* Kh = K  + ((size_t)bb * NUM_HEADS + h) * SD * HEAD_DIM;
  const h16* Vh = Vt + ((size_t)bb * NUM_HEADS + h) * HEAD_DIM * SD;

  v16h aq[2][2];
#pragma unroll
  for (int q = 0; q < 2; ++q) {
    const h16* Qrow = Qh + (size_t)(j0 + q * 16 + l15) * HEAD_DIM;
    aq[q][0] = ldfrag(Qrow, 0, half);
    aq[q][1] = ldfrag(Qrow, 32, half);
  }

  float m8[2][8], l8[2][8];
  v8f o[2][4] = {};
#pragma unroll
  for (int q = 0; q < 2; ++q)
#pragma unroll
    for (int r = 0; r < 8; ++r) { m8[q][r] = -1e30f; l8[q][r] = 0.f; }
  h16* myl = &lds[wib][0];              // 32 rows x 32 cols, row-major

  for (int kc = 0; kc < SEG; kc += 32) {
    const int kb = seg * SEG + kc;
    const h16* Kr0 = Kh + (size_t)(kb + l15) * HEAD_DIM;
    const h16* Kr1 = Kh + (size_t)(kb + 16 + l15) * HEAD_DIM;
    const v16h bk00 = ldfrag(Kr0, 0, half);
    const v16h bk01 = ldfrag(Kr0, 32, half);
    const v16h bk10 = ldfrag(Kr1, 0, half);
    const v16h bk11 = ldfrag(Kr1, 32, half);
    v8f c[2][2] = {};
#pragma unroll
    for (int q = 0; q < 2; ++q) {
      c[q][0] = wmma_f16(aq[q][0], bk00, c[q][0]);
      c[q][0] = wmma_f16(aq[q][1], bk01, c[q][0]);
      c[q][1] = wmma_f16(aq[q][0], bk10, c[q][1]);
      c[q][1] = wmma_f16(aq[q][1], bk11, c[q][1]);
    }

    // online softmax: rows live in vgpr index + 8*(lane/16); cols across the
    // 16-lane half, so xor-shuffles 8/4/2/1 reduce within each half.
#pragma unroll
    for (int q = 0; q < 2; ++q)
#pragma unroll
      for (int r = 0; r < 8; ++r) {
        const float s0 = c[q][0][r], s1 = c[q][1][r];
        float mx = fmaxf(s0, s1);
        mx = fmaxf(mx, __shfl_xor(mx, 8, 32));
        mx = fmaxf(mx, __shfl_xor(mx, 4, 32));
        mx = fmaxf(mx, __shfl_xor(mx, 2, 32));
        mx = fmaxf(mx, __shfl_xor(mx, 1, 32));
        const float mn   = fmaxf(m8[q][r], mx);
        const float corr = __expf(m8[q][r] - mn);
        const float p0   = __expf(s0 - mn);
        const float p1   = __expf(s1 - mn);
        float sm = p0 + p1;
        sm += __shfl_xor(sm, 8, 32);
        sm += __shfl_xor(sm, 4, 32);
        sm += __shfl_xor(sm, 2, 32);
        sm += __shfl_xor(sm, 1, 32);
        l8[q][r] = l8[q][r] * corr + sm;
        m8[q][r] = mn;
        o[q][0][r] *= corr; o[q][1][r] *= corr;
        o[q][2][r] *= corr; o[q][3][r] *= corr;
        const int row = q * 16 + r + 8 * half;
        myl[row * 32 + l15]      = (h16)p0;
        myl[row * 32 + 16 + l15] = (h16)p1;
      }
    asm volatile("s_wait_dscnt 0x0" ::: "memory");   // cross-lane LDS RAW
    v16h pf[2];
    pf[0] = ldfrag(myl + l15 * 32, 0, half);
    pf[1] = ldfrag(myl + (16 + l15) * 32, 0, half);
#pragma unroll
    for (int t = 0; t < 4; ++t) {
      const h16* Vr = Vh + (size_t)(t * 16 + l15) * SD + kb;
      const v16h bv = ldfrag(Vr, 0, half);
      o[0][t] = wmma_f16(pf[0], bv, o[0][t]);
      o[1][t] = wmma_f16(pf[1], bv, o[1][t]);
    }
  }

#pragma unroll
  for (int q = 0; q < 2; ++q)
#pragma unroll
    for (int t = 0; t < 4; ++t)
#pragma unroll
      for (int r = 0; r < 8; ++r) {
        const int j = j0 + q * 16 + r + 8 * half;
        const int d = t * 16 + l15;
        O[((size_t)bb * SD + j) * D_MODEL + h * HEAD_DIM + d] =
            (h16)(o[q][t][r] / l8[q][r]);
      }
}

// ---------------------------------------------------------------------------
// Kernel 4: output projection + scatter. One wave = 32x64 tile; block shares
// the 64 N-columns across its 8 waves (WGP$ B reuse).
// out[b, br+4j, n] = 0.25 * (O[bb,j,:] . Wo[br,n,:] + bo[br,n])
// ---------------------------------------------------------------------------
__global__ __launch_bounds__(256) void k_proj(const h16* __restrict__ O,
                                              const h16* __restrict__ wo,
                                              const float* __restrict__ bo,
                                              float* __restrict__ out) {
  const int lane = threadIdx.x & 31;
  const int l15  = lane & 15;
  const int half = lane >> 4;
  const int wib  = threadIdx.x >> 5;
  int gb = blockIdx.x;                    // 16 * 8 * 16 = 2048 blocks
  const int nt   = gb % 16; gb /= 16;     // 1024/64
  const int mblk = gb % 8;
  const int bb   = gb / 8;
  const int m0 = (mblk * 8 + wib) * 32;
  const int n0 = nt * 64;
  const int br = bb >> 2, b = bb & 3;

  const h16* Arow[2];
  Arow[0] = O + ((size_t)bb * SD + m0 + l15) * D_MODEL;
  Arow[1] = O + ((size_t)bb * SD + m0 + 16 + l15) * D_MODEL;
  const h16* W = wo + (size_t)br * D_MODEL * D_MODEL;
  const h16* Brow[4];
#pragma unroll
  for (int t = 0; t < 4; ++t)
    Brow[t] = W + (size_t)(n0 + t * 16 + l15) * D_MODEL;

  v8f acc[2][4] = {};
  for (int k = 0; k < D_MODEL; k += 32) {
    const v16h a0 = ldfrag(Arow[0], k, half);
    const v16h a1 = ldfrag(Arow[1], k, half);
#pragma unroll
    for (int t = 0; t < 4; ++t) {
      const v16h bt = ldfrag(Brow[t], k, half);
      acc[0][t] = wmma_f16(a0, bt, acc[0][t]);
      acc[1][t] = wmma_f16(a1, bt, acc[1][t]);
    }
  }

#pragma unroll
  for (int q = 0; q < 2; ++q)
#pragma unroll
    for (int t = 0; t < 4; ++t) {
      const int n = n0 + t * 16 + l15;
      const float bias = bo[br * D_MODEL + n];
#pragma unroll
      for (int r = 0; r < 8; ++r) {
        const int j = m0 + q * 16 + r + 8 * half;
        const int tseq = br + 4 * j;
        out[((size_t)b * SEQ + tseq) * D_MODEL + n] =
            0.25f * (acc[q][t][r] + bias);
      }
    }
}

// ---------------------------------------------------------------------------
extern "C" void kernel_launch(void* const* d_in, const int* in_sizes, int n_in,
                              void* d_out, int out_size, void* d_ws,
                              size_t ws_size, hipStream_t stream) {
  const float* x    = (const float*)d_in[0];
  const float* Wqkv = (const float*)d_in[1];
  const float* bqkv = (const float*)d_in[2];
  const float* Wo   = (const float*)d_in[3];
  const float* bo   = (const float*)d_in[4];
  float* out = (float*)d_out;

  char* ws = (char*)d_ws;
  size_t off = 0;
  h16* xbf = (h16*)(ws + off); off += (size_t)BATCH * SEQ * D_MODEL * 2;
  h16* wqb = (h16*)(ws + off); off += (size_t)DIL * 3 * D_MODEL * D_MODEL * 2;
  h16* wob = (h16*)(ws + off); off += (size_t)DIL * D_MODEL * D_MODEL * 2;
  const size_t qkvElems = (size_t)DIL * BATCH * NUM_HEADS * SD * HEAD_DIM;
  h16* Q  = (h16*)(ws + off); off += qkvElems * 2;
  h16* K  = (h16*)(ws + off); off += qkvElems * 2;
  h16* Vt = (h16*)(ws + off); off += qkvElems * 2;
  h16* O  = (h16*)(ws + off); off += (size_t)DIL * BATCH * SD * D_MODEL * 2;
  (void)off; (void)ws_size; (void)in_sizes; (void)n_in; (void)out_size;

  k_cvt<<<2048, 256, 0, stream>>>(x, xbf, (long)BATCH * SEQ * D_MODEL);
  k_cvt<<<1024, 256, 0, stream>>>(Wqkv, wqb, (long)DIL * 3 * D_MODEL * D_MODEL);
  k_cvt<<<512, 256, 0, stream>>>(Wo, wob, (long)DIL * D_MODEL * D_MODEL);

  // 48 nt * 8 mblk * 4 b * 4 br = 6144 blocks, 8 waves each
  k_qkv<<<6144, 256, 0, stream>>>(xbf, wqb, bqkv, Q, K, Vt);
  // 2 qb * 4 seg * 16 h * 16 bb = 2048 blocks
  k_attn<<<2048, 256, 0, stream>>>(Q, K, Vt, O);
  // 16 nt * 8 mblk * 16 bb = 2048 blocks
  k_proj<<<2048, 256, 0, stream>>>(O, wob, bo, out);
}
